// LongformerSelfAttentionForBart_14156212208083
// MI455X (gfx1250) — matrix operand order repaired and checked by reading, not compiled, not run on hardware
//
#include <hip/hip_runtime.h>

typedef unsigned short u16;
typedef unsigned int   u32;
typedef __attribute__((ext_vector_type(16))) _Float16 v16h;
typedef __attribute__((ext_vector_type(8)))  float    v8f;
typedef __attribute__((__vector_size__(4 * sizeof(int)))) int v4i;

union Frag {
  v16h v;
  u16  u[16];
  u32  d[8];
};

#define BB 2
#define SS 4096
#define DD 1024
#define HH 16
#define DH 64
#define WW 256
#define NCC 16

#define AS1 __attribute__((address_space(1)))
#define AS3 __attribute__((address_space(3)))

// ---- CDNA5 async global->LDS copy path (ASYNCcnt), with safe fallback ------
#if defined(__has_builtin)
#  if __has_builtin(__builtin_amdgcn_global_load_async_to_lds_b128)
#    define HAVE_ASYNC 1
#  else
#    define HAVE_ASYNC 0
#  endif
#  if __has_builtin(__builtin_amdgcn_s_wait_asynccnt)
#    define HAVE_ASYNC_WAIT 1
#  else
#    define HAVE_ASYNC_WAIT 0
#  endif
#else
#  define HAVE_ASYNC 0
#  define HAVE_ASYNC_WAIT 0
#endif

__device__ __forceinline__ void async_copy16(const void* g, void* l){
#if HAVE_ASYNC
  __builtin_amdgcn_global_load_async_to_lds_b128(
      (AS1 v4i*)(void*)g, (AS3 v4i*)l, 0, 0);
#else
  *(uint4*)l = *(const uint4*)g;
#endif
}

__device__ __forceinline__ void async_wait_all(){
#if HAVE_ASYNC
#  if HAVE_ASYNC_WAIT
  __builtin_amdgcn_s_wait_asynccnt(0);
#  else
  asm volatile("s_wait_asynccnt 0x0" ::: "memory");
#  endif
#endif
}

__device__ __forceinline__ u16 f2h(float f){
  union { _Float16 h; u16 u; } x;
  x.h = (_Float16)f;
  return x.u;
}

__device__ __forceinline__ v8f wmma16(v16h a, v16h b, v8f c){
  // D = A(16x32 f16) * B(32x16 f16) + C(16x16 f32)
  return __builtin_amdgcn_wmma_f32_16x16x32_f16(false, a, false, b, (short)0, c, false, false);
}

// ---------------------------------------------------------------------------
// fp32 -> fp16 conversion
// ---------------------------------------------------------------------------
__global__ __launch_bounds__(512) void cvt_f2h_kernel(const float* __restrict__ in,
                                                      u16* __restrict__ out, int n){
  int i = blockIdx.x * blockDim.x + threadIdx.x;
  if (i < n) out[i] = f2h(in[i]);
}

// ---------------------------------------------------------------------------
// QKV projection GEMM: (8192 x 1024) @ (1024 x 1024) + bias, scatter to
// head-major [B,H,S,DH] f16.  grid = (128, 16, 3), block = 128 (4 waves).
// ---------------------------------------------------------------------------
__global__ __launch_bounds__(128) void gemm_qkv_kernel(
    const u16* __restrict__ A,
    const u16* __restrict__ wq, const u16* __restrict__ wk, const u16* __restrict__ wv,
    const float* __restrict__ bq, const float* __restrict__ bk, const float* __restrict__ bv,
    u16* __restrict__ qo, u16* __restrict__ ko, u16* __restrict__ vo)
{
  __shared__ u16 As[64*40];   // [row][k]  (row stride 40 -> 80B, 16B aligned)
  __shared__ u16 Bs[64*40];   // [n][k]    (transposed for contiguous-k frag reads)

  const int tid = threadIdx.x;
  const int wid = tid >> 5, lane = tid & 31;
  const int lan16 = lane & 15, lgrp = lane >> 4;
  const int bm = blockIdx.x, bn = blockIdx.y, z = blockIdx.z;

  const u16*   Wm   = (z==0) ? wq : (z==1) ? wk : wv;
  const float* bias = (z==0) ? bq : (z==1) ? bk : bv;
  u16*         out  = (z==0) ? qo : (z==1) ? ko : vo;
  const float  scale = (z==0) ? 0.125f : 1.0f;   // q scaled by 1/sqrt(DH)

  const int wr = (wid >> 1) * 32, wc = (wid & 1) * 32;

  v8f acc[2][2];
  for (int i=0;i<2;++i) for (int j=0;j<2;++j) for (int r=0;r<8;++r) acc[i][j][r] = 0.0f;

  for (int k0 = 0; k0 < DD; k0 += 32){
    // ---- A tile 64x32: async global->LDS b128 copies
    #pragma unroll
    for (int p=0;p<2;++p){
      int seg = tid + p*128;              // 0..255
      int row = seg >> 2;                 // 0..63
      int ks  = (seg & 3) * 8;            // 0,8,16,24
      async_copy16(A + (size_t)(bm*64 + row)*DD + k0 + ks, &As[row*40 + ks]);
    }
    // ---- B tile 32x64, stored transposed [n][k] (sync scatter)
    #pragma unroll
    for (int p=0;p<2;++p){
      int seg = tid + p*128;
      int kk = seg >> 3;                  // 0..31
      int ns = (seg & 7) * 8;             // 0..56
      uint4 dB = *(const uint4*)(Wm + (size_t)(k0 + kk)*DD + bn*64 + ns);
      const u16* pv = (const u16*)&dB;
      #pragma unroll
      for (int j=0;j<8;++j) Bs[(ns+j)*40 + kk] = pv[j];
    }
    async_wait_all();
    __syncthreads();

    // ---- fragments in contiguous k order (k = 16*lgrp + s), same for A and B
    Frag a[2], bf[2];
    #pragma unroll
    for (int rt=0; rt<2; ++rt){
      const u16* ar = &As[(wr + rt*16 + lan16)*40];
      int base = lgrp*16;                 // lanes 16-31 hold K+16
      #pragma unroll
      for (int j=0;j<4;++j) a[rt].d[j]   = *(const u32*)(ar + base + 2*j);
      #pragma unroll
      for (int j=0;j<4;++j) a[rt].d[4+j] = *(const u32*)(ar + base + 8 + 2*j);
    }
    #pragma unroll
    for (int ct=0; ct<2; ++ct){
      const u16* br = &Bs[(wc + ct*16 + lan16)*40];
      int kb = lgrp*16;                   // lanes 16-31 hold K+16
      #pragma unroll
      for (int j=0;j<8;++j) bf[ct].d[j] = *(const u32*)(br + kb + 2*j);
    }
    #pragma unroll
    for (int rt=0; rt<2; ++rt)
      #pragma unroll
      for (int ct=0; ct<2; ++ct)
        acc[rt][ct] = wmma16(a[rt].v, bf[ct].v, acc[rt][ct]);
    __syncthreads();
  }

  // ---- epilogue: bias + scale, scatter to [B,H,S,DH] f16
  #pragma unroll
  for (int rt=0; rt<2; ++rt){
    #pragma unroll
    for (int ct=0; ct<2; ++ct){
      int gn = bn*64 + wc + ct*16 + lan16;
      float bb = bias[gn];
      int hsel = gn >> 6, dh = gn & 63;
      #pragma unroll
      for (int r=0; r<8; ++r){
        int gm = bm*64 + wr + rt*16 + r + lgrp*8;
        int bsel = gm >> 12, s = gm & (SS-1);
        float val = (acc[rt][ct][r] + bb) * scale;
        out[((size_t)(bsel*HH + hsel)*SS + s)*DH + dh] = f2h(val);
      }
    }
  }
}

// ---------------------------------------------------------------------------
// Output projection GEMM: ctx(8192x1024 f16) @ Wo + bo -> d_out f32 row-major
// ---------------------------------------------------------------------------
__global__ __launch_bounds__(128) void gemm_out_kernel(
    const u16* __restrict__ A, const u16* __restrict__ Wm,
    const float* __restrict__ bias, float* __restrict__ out)
{
  __shared__ u16 As[64*40];
  __shared__ u16 Bs[64*40];

  const int tid = threadIdx.x;
  const int wid = tid >> 5, lane = tid & 31;
  const int lan16 = lane & 15, lgrp = lane >> 4;
  const int bm = blockIdx.x, bn = blockIdx.y;
  const int wr = (wid >> 1) * 32, wc = (wid & 1) * 32;

  v8f acc[2][2];
  for (int i=0;i<2;++i) for (int j=0;j<2;++j) for (int r=0;r<8;++r) acc[i][j][r] = 0.0f;

  for (int k0 = 0; k0 < DD; k0 += 32){
    #pragma unroll
    for (int p=0;p<2;++p){
      int seg = tid + p*128;
      int row = seg >> 2;
      int ks  = (seg & 3) * 8;
      async_copy16(A + (size_t)(bm*64 + row)*DD + k0 + ks, &As[row*40 + ks]);
    }
    #pragma unroll
    for (int p=0;p<2;++p){
      int seg = tid + p*128;
      int kk = seg >> 3;
      int ns = (seg & 7) * 8;
      uint4 dB = *(const uint4*)(Wm + (size_t)(k0 + kk)*DD + bn*64 + ns);
      const u16* pv = (const u16*)&dB;
      #pragma unroll
      for (int j=0;j<8;++j) Bs[(ns+j)*40 + kk] = pv[j];
    }
    async_wait_all();
    __syncthreads();

    Frag a[2], bf[2];
    #pragma unroll
    for (int rt=0; rt<2; ++rt){
      const u16* ar = &As[(wr + rt*16 + lan16)*40];
      int base = lgrp*16;
      #pragma unroll
      for (int j=0;j<4;++j) a[rt].d[j]   = *(const u32*)(ar + base + 2*j);
      #pragma unroll
      for (int j=0;j<4;++j) a[rt].d[4+j] = *(const u32*)(ar + base + 8 + 2*j);
    }
    #pragma unroll
    for (int ct=0; ct<2; ++ct){
      const u16* br = &Bs[(wc + ct*16 + lan16)*40];
      int kb = lgrp*16;
      #pragma unroll
      for (int j=0;j<8;++j) bf[ct].d[j] = *(const u32*)(br + kb + 2*j);
    }
    #pragma unroll
    for (int rt=0; rt<2; ++rt)
      #pragma unroll
      for (int ct=0; ct<2; ++ct)
        acc[rt][ct] = wmma16(a[rt].v, bf[ct].v, acc[rt][ct]);
    __syncthreads();
  }

  #pragma unroll
  for (int rt=0; rt<2; ++rt){
    #pragma unroll
    for (int ct=0; ct<2; ++ct){
      int gn = bn*64 + wc + ct*16 + lan16;
      float bb = bias[gn];
      #pragma unroll
      for (int r=0; r<8; ++r){
        int gm = bm*64 + wr + rt*16 + r + lgrp*8;
        out[(size_t)gm*DD + gn] = acc[rt][ct][r] + bb;
      }
    }
  }
}

// ---------------------------------------------------------------------------
// Banded (Longformer) attention, flash-style online softmax.
// grid = B*H*NC*2 = 1024 blocks, block = 256 (8 waves).
// Block cooperatively stages each 32-key K/V tile in LDS (K via async copies);
// each wave owns 16 queries and consumes tiles intersecting its band.
// Out-of-range keys are CLAMPED: every such position is index-masked to -1e9,
// so its softmax weight is exactly 0 (transient fully-masked tiles are
// annihilated by the online rescale), making clamped data irrelevant.
// ---------------------------------------------------------------------------
__global__ __launch_bounds__(256) void attn_kernel(
    const u16* __restrict__ qb, const u16* __restrict__ kb, const u16* __restrict__ vb,
    const float* __restrict__ amask, u16* __restrict__ ctxb)
{
  __shared__ u16 Ks[32*72];           // [key][dh]   stride 72 (144B rows, 16B aligned)
  __shared__ u16 Vs[64*40];           // [dh][key]   stride 40 (transposed)
  __shared__ u16 Pbuf[8 * 16 * 34];   // per-wave P tile, stride 34 (conflict-free)

  const int tid = threadIdx.x;
  const int wid = tid >> 5, lane = tid & 31;
  const int lan16 = lane & 15, lgrp = lane >> 4;

  const int bidx = blockIdx.x;
  const int hb   = bidx & 1;          // which half of the chunk's 16 q-tiles
  const int cidx = bidx >> 1;         // 0..511
  const int c = cidx & 15;
  const int h = (cidx >> 4) & 15;
  const int b = cidx >> 8;

  const int q0 = hb*128 + wid*16;     // this wave's query base (local to chunk)

  const u16* qptr = qb + (size_t)(b*HH + h)*SS*DH;
  const u16* kptr = kb + (size_t)(b*HH + h)*SS*DH;
  const u16* vptr = vb + (size_t)(b*HH + h)*SS*DH;

  // Q A-fragments (dh 0..31 and 32..63), held in registers across key loop
  Frag aq[2];
  {
    const u16* qr = qptr + (size_t)(c*WW + q0 + lan16)*DH;
    #pragma unroll
    for (int t=0;t<2;++t){
      int base = t*32 + lgrp*16;
      #pragma unroll
      for (int j=0;j<4;++j) aq[t].d[j]   = *(const u32*)(qr + base + 2*j);
      #pragma unroll
      for (int j=0;j<4;++j) aq[t].d[4+j] = *(const u32*)(qr + base + 8 + 2*j);
    }
  }

  float Mrow[8], Lrow[8];
  v8f o[4];
  #pragma unroll
  for (int r=0;r<8;++r){ Mrow[r] = -1e38f; Lrow[r] = 0.0f; }
  #pragma unroll
  for (int dt=0;dt<4;++dt)
    #pragma unroll
    for (int r=0;r<8;++r) o[dt][r] = 0.0f;

  // per-wave band and per-block (union) key-tile range
  const int lo_w = q0 >> 5;
  int hi_w = (q0 + 15 + 512) >> 5;  if (hi_w > 23) hi_w = 23;
  const int kt_blo = hb * 4;
  const int kt_bhi = (hb == 0) ? 19 : 23;

  u16* P = Pbuf + wid * 16 * 34;

  for (int kt = kt_blo; kt <= kt_bhi; ++kt){
    const int kl0   = kt * 32;              // key_local base
    const int gbase = c*WW - WW + kl0;      // global key index of n=0

    __syncthreads();                        // previous tile fully consumed
    // ---- stage K [key][dh] via async copies; V transposed [dh][key] sync
    {
      int key = tid >> 3;                   // 0..31
      int dhs = (tid & 7) * 8;              // 0,8,..,56
      int g = gbase + key;
      int gc = g < 0 ? 0 : (g >= SS ? SS-1 : g);
      async_copy16(kptr + (size_t)gc*DH + dhs, &Ks[key*72 + dhs]);
      uint4 dV = *(const uint4*)(vptr + (size_t)gc*DH + dhs);
      const u16* pv = (const u16*)&dV;
      #pragma unroll
      for (int j=0;j<8;++j) Vs[(dhs+j)*40 + key] = pv[j];
    }
    async_wait_all();
    __syncthreads();

    if (kt < lo_w || kt > hi_w) continue;   // outside this wave's band

    // ---- K B-fragments from LDS: [ct = key subtile][dt = dh half]
    Frag bkf[2][2];
    float mval[2];
    #pragma unroll
    for (int ct=0; ct<2; ++ct){
      int g = gbase + ct*16 + lan16;
      bool valid = (g >= 0) && (g < SS);
      mval[ct] = valid ? amask[b*SS + g] : 0.0f;
      const u16* kr = &Ks[(ct*16 + lan16)*72];
      #pragma unroll
      for (int dt=0; dt<2; ++dt){
        int base = dt*32 + lgrp*16;
        #pragma unroll
        for (int j=0;j<8;++j) bkf[ct][dt].d[j] = *(const u32*)(kr + base + 2*j);
      }
    }

    // ---- scores: 16q x 32k (two 16x16 C-frags), K-dim = DH = 2 WMMA steps
    v8f sc[2];
    #pragma unroll
    for (int ct=0; ct<2; ++ct){
      v8f z;
      #pragma unroll
      for (int r=0;r<8;++r) z[r] = 0.0f;
      z = wmma16(aq[0].v, bkf[ct][0].v, z);
      z = wmma16(aq[1].v, bkf[ct][1].v, z);
      sc[ct] = z;
    }

    // ---- mask + running row max (reduce across 16 lanes of each row)
    float ex[8], nM[8];
    #pragma unroll
    for (int r=0;r<8;++r){
      int ql = q0 + r + lgrp*8;             // query local index (0..255)
      {
        int kl = kl0 + lan16;  int g = gbase + lan16;
        int rel = kl - ql;
        bool ok = (rel >= 0) && (rel <= 512) && (g >= 0) && (g < SS);
        sc[0][r] = ok ? (sc[0][r] + mval[0]) : -1.0e9f;
      }
      {
        int kl = kl0 + 16 + lan16;  int g = gbase + 16 + lan16;
        int rel = kl - ql;
        bool ok = (rel >= 0) && (rel <= 512) && (g >= 0) && (g < SS);
        sc[1][r] = ok ? (sc[1][r] + mval[1]) : -1.0e9f;
      }
      float t = fmaxf(sc[0][r], sc[1][r]);
      #pragma unroll
      for (int off=8; off>=1; off>>=1) t = fmaxf(t, __shfl_xor(t, off, 32));
      float newM = fmaxf(Mrow[r], t);
      ex[r] = __expf(Mrow[r] - newM);
      nM[r] = newM;
    }

    // ---- P = exp(s - M), row sums, online-softmax state update
    #pragma unroll
    for (int r=0;r<8;++r){
      float p0 = __expf(sc[0][r] - nM[r]);
      float p1 = __expf(sc[1][r] - nM[r]);
      sc[0][r] = p0;  sc[1][r] = p1;
      float s = p0 + p1;
      #pragma unroll
      for (int off=8; off>=1; off>>=1) s += __shfl_xor(s, off, 32);
      Lrow[r] = Lrow[r]*ex[r] + s;
      Mrow[r] = nM[r];
    }
    #pragma unroll
    for (int dt=0;dt<4;++dt)
      #pragma unroll
      for (int r=0;r<8;++r) o[dt][r] *= ex[r];

    // ---- stage P (16x32 f16) through this wave's LDS slot, re-read as A-frag
    #pragma unroll
    for (int r=0;r<8;++r){
      int row = r + lgrp*8;
      P[row*34 + lan16]      = f2h(sc[0][r]);
      P[row*34 + 16 + lan16] = f2h(sc[1][r]);
    }
    Frag aP;
    {
      const u16* pr = P + lan16*34;
      int base = lgrp*16;
      #pragma unroll
      for (int j=0;j<4;++j) aP.d[j]   = *(const u32*)(pr + base + 2*j);
      #pragma unroll
      for (int j=0;j<4;++j) aP.d[4+j] = *(const u32*)(pr + base + 8 + 2*j);
    }

    // ---- O += P @ V  (4 dh subtiles of 16), V frags from transposed LDS
    #pragma unroll
    for (int dt=0; dt<4; ++dt){
      Frag bvf;
      const u16* vr = &Vs[(dt*16 + lan16)*40];
      int kbse = lgrp*16;
      #pragma unroll
      for (int j=0;j<8;++j) bvf.d[j] = *(const u32*)(vr + kbse + 2*j);
      o[dt] = wmma16(aP.v, bvf.v, o[dt]);
    }
  }

  // ---- normalize and store ctx in [B,S,D] f16 (D = h*64 + dh)
  #pragma unroll
  for (int dt=0; dt<4; ++dt){
    #pragma unroll
    for (int r=0; r<8; ++r){
      float val = o[dt][r] / Lrow[r];
      int sg  = c*WW + q0 + r + lgrp*8;
      int col = h*DH + dt*16 + lan16;
      ctxb[((size_t)b*SS + sg)*DD + col] = f2h(val);
    }
  }
}

// ---------------------------------------------------------------------------
// launch
// ---------------------------------------------------------------------------
extern "C" void kernel_launch(void* const* d_in, const int* in_sizes, int n_in,
                              void* d_out, int out_size, void* d_ws, size_t ws_size,
                              hipStream_t stream)
{
  (void)in_sizes; (void)n_in; (void)out_size; (void)ws_size;
  const float* hs    = (const float*)d_in[0];
  const float* amask = (const float*)d_in[1];
  const float* Wq = (const float*)d_in[2];  const float* bq = (const float*)d_in[3];
  const float* Wk = (const float*)d_in[4];  const float* bk = (const float*)d_in[5];
  const float* Wv = (const float*)d_in[6];  const float* bv = (const float*)d_in[7];
  const float* Wo = (const float*)d_in[8];  const float* bo = (const float*)d_in[9];
  float* outp = (float*)d_out;

  char* ws = (char*)d_ws;
  size_t off = 0;
  auto wsalloc = [&](size_t bytes) -> void* {
    void* p = ws + off;
    off += (bytes + 255) & ~(size_t)255;
    return p;
  };
  const size_t nHS = (size_t)BB*SS*DD;       // 8388608
  const size_t nW  = (size_t)DD*DD;          // 1048576

  u16* hsb  = (u16*)wsalloc(nHS*2);
  u16* wqb  = (u16*)wsalloc(nW*2);
  u16* wkb  = (u16*)wsalloc(nW*2);
  u16* wvb  = (u16*)wsalloc(nW*2);
  u16* wob  = (u16*)wsalloc(nW*2);
  u16* qbuf = (u16*)wsalloc(nHS*2);
  u16* kbuf = (u16*)wsalloc(nHS*2);
  u16* vbuf = (u16*)wsalloc(nHS*2);
  u16* ctxb = (u16*)wsalloc(nHS*2);

  cvt_f2h_kernel<<<(int)(nHS/512), 512, 0, stream>>>(hs, hsb, (int)nHS);
  cvt_f2h_kernel<<<(int)(nW/512),  512, 0, stream>>>(Wq, wqb, (int)nW);
  cvt_f2h_kernel<<<(int)(nW/512),  512, 0, stream>>>(Wk, wkb, (int)nW);
  cvt_f2h_kernel<<<(int)(nW/512),  512, 0, stream>>>(Wv, wvb, (int)nW);
  cvt_f2h_kernel<<<(int)(nW/512),  512, 0, stream>>>(Wo, wob, (int)nW);

  gemm_qkv_kernel<<<dim3(128, 16, 3), 128, 0, stream>>>(
      hsb, wqb, wkb, wvb, bq, bk, bv, qbuf, kbuf, vbuf);

  attn_kernel<<<BB*HH*NCC*2, 256, 0, stream>>>(qbuf, kbuf, vbuf, amask, ctxb);

  gemm_out_kernel<<<dim3(128, 16), 128, 0, stream>>>(ctxb, wob, bo, outp);
}
